// EGNNEncoder_54863912239508
// MI455X (gfx1250) — matrix operand order repaired and checked by reading, not compiled, not run on hardware
//
#include <hip/hip_runtime.h>
#include <hip/hip_bf16.h>

// ---------------------------------------------------------------------------
// EGNN encoder for MI455X (gfx1250): wave32, v_wmma_f32_16x16x32_f16,
// TDM (tensor_load_to_lds) weight staging, LDS B-fragments, fused scatter.
// N=50000 nodes, E=250000 edges, D=128, F=12, L=4.
// ---------------------------------------------------------------------------

#define NN 50000
#define NE 250000
#define DD 128
#define FF 12
#define LL 4

typedef __attribute__((ext_vector_type(16))) _Float16 v16h;
typedef __attribute__((ext_vector_type(8)))  _Float16 v8h;
typedef __attribute__((ext_vector_type(8)))  float    v8f;
typedef __attribute__((ext_vector_type(4)))  unsigned int u32x4;
typedef __attribute__((ext_vector_type(8)))  int          i32x8;
typedef __attribute__((ext_vector_type(4)))  int          i32x4;

__device__ __forceinline__ float silu_f(float v) { return v / (1.0f + __expf(-v)); }

// ---------------------------------------------------------------------------
// TDM: DMA one contiguous blob (nelem8 x 8 bytes) from global to LDS offset.
// Descriptor per CDNA5 ISA (2D, 1 row, data_size=8B).
// 6-arg builtin form (clang-23 / therock-10.0 lane).
// ---------------------------------------------------------------------------
__device__ __forceinline__ void tdm_load_blob(const void* gptr, unsigned lds_off,
                                              unsigned nelem8) {
  const unsigned long long ga = (unsigned long long)(uintptr_t)gptr;
  u32x4 g0;
  g0[0] = 1u;                                        // count=1, user mode
  g0[1] = lds_off;                                   // lds_addr (bytes)
  g0[2] = (unsigned)(ga & 0xffffffffu);              // global_addr[31:0]
  g0[3] = (unsigned)((ga >> 32) & 0x01ffffffu) | (2u << 30);  // ga[56:32], type=2
  i32x8 g1;
  g1[0] = (int)(3u << 16);                           // data_size = 8B
  g1[1] = (int)(nelem8 << 16);                       // tensor_dim0[15:0]
  g1[2] = (int)(1u << 16);                           // tensor_dim1 = 1
  g1[3] = (int)(nelem8 << 16);                       // tile_dim0
  g1[4] = 1;                                         // tile_dim1 = 1
  g1[5] = (int)nelem8;                               // tensor_dim0_stride
  g1[6] = 0;
  g1[7] = 0;
  i32x4 z4 = {0, 0, 0, 0};
  i32x8 z8 = {0, 0, 0, 0, 0, 0, 0, 0};
  __builtin_amdgcn_tensor_load_to_lds(g0, g1, z4, z4, z8, 0);
}

// A-fragment (16x32 f16, MxK) from one fp32 row of a row-major matrix.
// Lane holds row M = lane&15; element i -> K = (i&7) + (i&8)*2 + 8*(lane>=16).
__device__ __forceinline__ v16h a_frag_f32(const float* __restrict__ rowp, int c0, int lane) {
  const int r0 = (lane & 16) >> 1;
  const float4 a0 = *(const float4*)(rowp + c0 + r0);
  const float4 a1 = *(const float4*)(rowp + c0 + r0 + 4);
  const float4 b0 = *(const float4*)(rowp + c0 + r0 + 16);
  const float4 b1 = *(const float4*)(rowp + c0 + r0 + 20);
  v16h f;
  f[0]  = (_Float16)a0.x; f[1]  = (_Float16)a0.y; f[2]  = (_Float16)a0.z; f[3]  = (_Float16)a0.w;
  f[4]  = (_Float16)a1.x; f[5]  = (_Float16)a1.y; f[6]  = (_Float16)a1.z; f[7]  = (_Float16)a1.w;
  f[8]  = (_Float16)b0.x; f[9]  = (_Float16)b0.y; f[10] = (_Float16)b0.z; f[11] = (_Float16)b0.w;
  f[12] = (_Float16)b1.x; f[13] = (_Float16)b1.y; f[14] = (_Float16)b1.z; f[15] = (_Float16)b1.w;
  return f;
}

// A-fragment from an f16 row in LDS.
__device__ __forceinline__ v16h a_frag_lds(const _Float16* __restrict__ rowp, int c0, int lane) {
  const int r0 = (lane & 16) >> 1;
  v8h lo = *(const v8h*)(rowp + c0 + r0);
  v8h hi = *(const v8h*)(rowp + c0 + r0 + 16);
  v16h f;
#pragma unroll
  for (int i = 0; i < 8; ++i) { f[i] = lo[i]; f[8 + i] = hi[i]; }
  return f;
}

// ---------------------------------------------------------------------------
// Pack [kchunks*32 x 128] fp32 weights into WMMA B-fragment order (f16):
// dst[((kc*8 + nt)*32 + lane)*16 + i] = W[kc*32 + (i&7) + (i&8)*2 + (lane&16)/2][nt*16 + (lane&15)]
// ---------------------------------------------------------------------------
__global__ void egnn_pack_kernel(const float* __restrict__ src, _Float16* __restrict__ dst,
                                 int kchunks) {
  int gid = blockIdx.x * blockDim.x + threadIdx.x;
  if (gid >= kchunks * 4096) return;
  int kc   = gid >> 12;
  int rem  = gid & 4095;
  int nt   = rem >> 9;
  int rem2 = rem & 511;
  int lane = rem2 >> 4;
  int i    = rem2 & 15;
  dst[gid] = (_Float16)src[(kc * 32 + (i & 7) + ((i & 8) << 1) + ((lane & 16) >> 1)) * DD + nt * 16 + (lane & 15)];
}

__global__ void egnn_zero_kernel(float* __restrict__ p, int count) {
  int i = blockIdx.x * blockDim.x + threadIdx.x;
  if (i < count) p[i] = 0.0f;
}
__global__ void egnn_copy_kernel(float* __restrict__ dst, const float* __restrict__ src, int count) {
  int i = blockIdx.x * blockDim.x + threadIdx.x;
  if (i < count) dst[i] = src[i];
}
__global__ void egnn_axpy_kernel(float* __restrict__ dst, const float* __restrict__ src, int count) {
  int i = blockIdx.x * blockDim.x + threadIdx.x;
  if (i < count) dst[i] += src[i];
}

// x = node_features @ node_proj_w + node_proj_b  (K=12, pure VALU)
__global__ __launch_bounds__(128) void egnn_proj_kernel(
    const float* __restrict__ nf, const float* __restrict__ W,
    const float* __restrict__ b, float* __restrict__ x) {
  __shared__ float s_nf[FF];
  const int n = blockIdx.x, t = threadIdx.x;
  if (t < FF) s_nf[t] = nf[n * FF + t];
  __syncthreads();
  float acc = b[t];
#pragma unroll
  for (int f = 0; f < FF; ++f) acc += s_nf[f] * W[f * DD + t];
  x[(size_t)n * DD + t] = acc;
}

// ---------------------------------------------------------------------------
// Edge kernel: block = 64 edges, 4 waves; wave w owns M-tile w (16 edges),
// all 128 output columns (8 accumulators). Weights (W1|W2 pack, 96KB) staged
// by TDM into LDS. msg stays in registers: scatter + pos-MLP reduce directly.
// Dynamic LDS: [0,98304) weights | [98304,+17408) h tiles | edge metadata.
// ---------------------------------------------------------------------------
__global__ __launch_bounds__(128) void egnn_edge_kernel(
    const float* __restrict__ x, const float* __restrict__ p,
    const int* __restrict__ ei,
    const _Float16* __restrict__ wep,   // packed W1 (8 kchunks) | W2 (4 kchunks)
    const float* __restrict__ w1last, const float* __restrict__ b1,
    const float* __restrict__ b2,
    const float* __restrict__ pw1, const float* __restrict__ pb1,
    const float* __restrict__ pw2, const float* __restrict__ pb2,
    float* __restrict__ agg, float* __restrict__ pacc) {
  extern __shared__ char smem[];
  _Float16* w1_lds = (_Float16*)smem;                    // 32768 halfs
  _Float16* w2_lds = (_Float16*)(smem + 65536);          // 16384 halfs
  _Float16* h_all  = (_Float16*)(smem + 98304);          // 4*16*136 halfs
  int*   s_src  = (int*)(smem + 98304 + 17408);
  int*   s_tgt  = s_src + 64;
  float* s_dist = (float*)(s_tgt + 64);
  float* s_ux   = s_dist + 64;
  float* s_uy   = s_ux + 64;

  const int tid  = threadIdx.x;
  const int wave = tid >> 5;
  const int lane = tid & 31;
  const int e0b  = blockIdx.x * 64;

  if (tid < 64) {
    int e = e0b + tid; if (e >= NE) e = NE - 1;
    const int s = ei[e];
    const int t = ei[NE + e];
    s_src[tid] = s; s_tgt[tid] = t;
    const float dx = p[2 * t] - p[2 * s];
    const float dy = p[2 * t + 1] - p[2 * s + 1];
    const float d  = sqrtf(dx * dx + dy * dy);
    s_dist[tid] = d;
    const float inv = 1.0f / (d + 1e-6f);
    s_ux[tid] = dx * inv; s_uy[tid] = dy * inv;
  }
  if (wave == 0) {                    // one TDM per block stages 96KB
    tdm_load_blob(wep, 0u, 12288u);
    __builtin_amdgcn_s_wait_tensorcnt(0);
  }
  __syncthreads();

  const int m    = lane & 15;
  const int em_a = wave * 16 + m;     // A-fragment row (edge slot 0..63)
  _Float16* h = h_all + wave * (16 * 136);

  // GEMM1: [16 x 257] @ [257 x 128] per wave (K=256 WMMA + exact fp32 dist row)
  v8f acc[8] = {};
#pragma unroll
  for (int kc = 0; kc < 8; ++kc) {
    const int node = (kc < 4) ? s_src[em_a] : s_tgt[em_a];
    const v16h a = a_frag_f32(x + (size_t)node * DD, (kc & 3) * 32, lane);
#pragma unroll
    for (int nt = 0; nt < 8; ++nt) {
      const v16h b = *(const v16h*)(w1_lds + ((kc * 8 + nt) * 32 + lane) * 16);
      acc[nt] = __builtin_amdgcn_wmma_f32_16x16x32_f16(false, a, false, b, (short)0, acc[nt], false, false);
    }
  }
  const int mb = (lane & 16) >> 1;
#pragma unroll
  for (int nt = 0; nt < 8; ++nt) {
    const int n = nt * 16 + (lane & 15);
    const float bb = b1[n], wl = w1last[n];
#pragma unroll
    for (int r = 0; r < 8; ++r) {
      const int em = wave * 16 + mb + r;
      h[(mb + r) * 136 + n] = (_Float16)silu_f(acc[nt][r] + bb + s_dist[em] * wl);
    }
  }
  // no barrier: h tile is wave-private

  // GEMM2: [16 x 128] @ [128 x 128]
  v8f acc2[8] = {};
#pragma unroll
  for (int kc = 0; kc < 4; ++kc) {
    const v16h a = a_frag_lds(h + m * 136, kc * 32, lane);
#pragma unroll
    for (int nt = 0; nt < 8; ++nt) {
      const v16h b = *(const v16h*)(w2_lds + ((kc * 8 + nt) * 32 + lane) * 16);
      acc2[nt] = __builtin_amdgcn_wmma_f32_16x16x32_f16(false, a, false, b, (short)0, acc2[nt], false, false);
    }
  }

  // Epilogue: msg = acc2 + b2; scatter into agg[tgt] and reduce for pos-MLP.
  float part[8] = {};
#pragma unroll
  for (int nt = 0; nt < 8; ++nt) {
    const int n = nt * 16 + (lane & 15);
    const float bb = b2[n], pwn = pw1[n];
#pragma unroll
    for (int r = 0; r < 8; ++r) {
      const int em  = wave * 16 + mb + r;
      const float msg = acc2[nt][r] + bb;
      part[r] += msg * pwn;
      if (e0b + em < NE)
        unsafeAtomicAdd(&agg[(size_t)s_tgt[em] * DD + n], msg);
    }
  }
  // Butterfly-reduce part[] over the 16 lanes that share each edge row.
#pragma unroll
  for (int r = 0; r < 8; ++r) {
    float v = part[r];
    v += __shfl_xor(v, 1, 16);
    v += __shfl_xor(v, 2, 16);
    v += __shfl_xor(v, 4, 16);
    v += __shfl_xor(v, 8, 16);
    part[r] = v;
  }
  if ((lane & 15) == 0) {             // lanes 0 and 16: 8 edges each
    const float w1d = pw1[DD], bp1 = pb1[0], w2s = pw2[0], bp2 = pb2[0];
#pragma unroll
    for (int r = 0; r < 8; ++r) {
      const int em = wave * 16 + mb + r;
      if (e0b + em < NE) {
        const float pw = silu_f(part[r] + s_dist[em] * w1d + bp1) * w2s + bp2;
        const int t = s_tgt[em];
        unsafeAtomicAdd(&pacc[2 * t],     s_ux[em] * pw);
        unsafeAtomicAdd(&pacc[2 * t + 1], s_uy[em] * pw);
      }
    }
  }
}

// ---------------------------------------------------------------------------
// Node kernel: block = 64 nodes, wave per 16-node tile, x += MLP([x, agg]).
// Dynamic LDS: [0,98304) weights | [98304,+17408) h tiles.
// ---------------------------------------------------------------------------
__global__ __launch_bounds__(128) void egnn_node_kernel(
    float* __restrict__ x, const float* __restrict__ agg,
    const _Float16* __restrict__ wnp,   // packed W1 (8 kc) | W2 (4 kc)
    const float* __restrict__ b1, const float* __restrict__ b2) {
  extern __shared__ char smem[];
  _Float16* w1_lds = (_Float16*)smem;
  _Float16* w2_lds = (_Float16*)(smem + 65536);
  _Float16* h_all  = (_Float16*)(smem + 98304);

  const int tid  = threadIdx.x;
  const int wave = tid >> 5;
  const int lane = tid & 31;
  const int n0b  = blockIdx.x * 64;

  if (wave == 0) {
    tdm_load_blob(wnp, 0u, 12288u);
    __builtin_amdgcn_s_wait_tensorcnt(0);
  }
  __syncthreads();

  const int m = lane & 15;
  int arow = n0b + wave * 16 + m; if (arow >= NN) arow = NN - 1;
  const size_t row = (size_t)arow * DD;
  _Float16* h = h_all + wave * (16 * 136);

  v8f acc[8] = {};
#pragma unroll
  for (int kc = 0; kc < 8; ++kc) {
    const float* src = (kc < 4) ? (x + row) : (agg + row);
    const v16h a = a_frag_f32(src, (kc & 3) * 32, lane);
#pragma unroll
    for (int nt = 0; nt < 8; ++nt) {
      const v16h b = *(const v16h*)(w1_lds + ((kc * 8 + nt) * 32 + lane) * 16);
      acc[nt] = __builtin_amdgcn_wmma_f32_16x16x32_f16(false, a, false, b, (short)0, acc[nt], false, false);
    }
  }
  const int mb = (lane & 16) >> 1;
#pragma unroll
  for (int nt = 0; nt < 8; ++nt) {
    const int n = nt * 16 + (lane & 15);
    const float bb = b1[n];
#pragma unroll
    for (int r = 0; r < 8; ++r)
      h[(mb + r) * 136 + n] = (_Float16)silu_f(acc[nt][r] + bb);
  }

  v8f acc2[8] = {};
#pragma unroll
  for (int kc = 0; kc < 4; ++kc) {
    const v16h a = a_frag_lds(h + m * 136, kc * 32, lane);
#pragma unroll
    for (int nt = 0; nt < 8; ++nt) {
      const v16h b = *(const v16h*)(w2_lds + ((kc * 8 + nt) * 32 + lane) * 16);
      acc2[nt] = __builtin_amdgcn_wmma_f32_16x16x32_f16(false, a, false, b, (short)0, acc2[nt], false, false);
    }
  }
#pragma unroll
  for (int nt = 0; nt < 8; ++nt) {
    const int n = nt * 16 + (lane & 15);
    const float bb = b2[n];
#pragma unroll
    for (int r = 0; r < 8; ++r) {
      const int node = n0b + wave * 16 + mb + r;
      if (node < NN) x[(size_t)node * DD + n] += acc2[nt][r] + bb;
    }
  }
}

// LayerNorm over D=128; one block per node.
__global__ __launch_bounds__(128) void egnn_ln_kernel(
    const float* __restrict__ x, const float* __restrict__ g,
    const float* __restrict__ b, float* __restrict__ out) {
  __shared__ float red[128];
  const int n = blockIdx.x, t = threadIdx.x;
  const float v = x[(size_t)n * DD + t];
  red[t] = v; __syncthreads();
#pragma unroll
  for (int s = 64; s > 0; s >>= 1) { if (t < s) red[t] += red[t + s]; __syncthreads(); }
  const float mu = red[0] / 128.0f;
  __syncthreads();
  const float dv = v - mu;
  red[t] = dv * dv; __syncthreads();
#pragma unroll
  for (int s = 64; s > 0; s >>= 1) { if (t < s) red[t] += red[t + s]; __syncthreads(); }
  const float var = red[0] / 128.0f;
  out[(size_t)n * DD + t] = dv * rsqrtf(var + 1e-5f) * g[t] + b[t];
}

// ---------------------------------------------------------------------------
extern "C" void kernel_launch(void* const* d_in, const int* in_sizes, int n_in,
                              void* d_out, int out_size, void* d_ws, size_t ws_size,
                              hipStream_t stream) {
  const float* node_features = (const float*)d_in[0];
  const float* pos           = (const float*)d_in[1];
  // d_in[2] edge_attr_raw, d_in[5..6] edge_proj: mathematically dead -> skipped
  const float* node_proj_w = (const float*)d_in[3];
  const float* node_proj_b = (const float*)d_in[4];
  const float* edge_w1 = (const float*)d_in[7];
  const float* edge_b1 = (const float*)d_in[8];
  const float* edge_w2 = (const float*)d_in[9];
  const float* edge_b2 = (const float*)d_in[10];
  const float* node_w1 = (const float*)d_in[11];
  const float* node_b1 = (const float*)d_in[12];
  const float* node_w2 = (const float*)d_in[13];
  const float* node_b2 = (const float*)d_in[14];
  const float* pos_w1  = (const float*)d_in[15];
  const float* pos_b1  = (const float*)d_in[16];
  const float* pos_w2  = (const float*)d_in[17];
  const float* pos_b2  = (const float*)d_in[18];
  const float* ln_g    = (const float*)d_in[19];
  const float* ln_b    = (const float*)d_in[20];
  const int*   ei      = (const int*)d_in[21];

  char* ws = (char*)d_ws;
  size_t off = 0;
  auto walloc = [&](size_t bytes) -> void* {
    void* p = ws + off;
    off = (off + bytes + 255) & ~(size_t)255;
    return p;
  };
  float* xbuf = (float*)walloc((size_t)NN * DD * 4);
  float* agg  = (float*)walloc((size_t)NN * DD * 4);
  float* pcur = (float*)walloc((size_t)NN * 2 * 4);
  float* pacc = (float*)walloc((size_t)NN * 2 * 4);
  _Float16* wep[LL]; _Float16* wnp[LL]; float* w1last[LL];
  for (int l = 0; l < LL; ++l) {
    wep[l] = (_Float16*)walloc(12 * 4096 * sizeof(_Float16));  // W1(8kc)|W2(4kc), 96KB
    wnp[l] = (_Float16*)walloc(12 * 4096 * sizeof(_Float16));
    w1last[l] = (float*)walloc(DD * 4);
  }

  // Pack weights into WMMA B-fragment order (contiguous W1|W2 per layer).
  for (int l = 0; l < LL; ++l) {
    egnn_pack_kernel<<<128, 256, 0, stream>>>(edge_w1 + (size_t)l * 257 * DD, wep[l], 8);
    egnn_pack_kernel<<<64, 256, 0, stream>>>(edge_w2 + (size_t)l * DD * DD, wep[l] + 8 * 4096, 4);
    egnn_copy_kernel<<<1, 128, 0, stream>>>(w1last[l], edge_w1 + (size_t)l * 257 * DD + 256 * DD, DD);
    egnn_pack_kernel<<<128, 256, 0, stream>>>(node_w1 + (size_t)l * 256 * DD, wnp[l], 8);
    egnn_pack_kernel<<<64, 256, 0, stream>>>(node_w2 + (size_t)l * DD * DD, wnp[l] + 8 * 4096, 4);
  }

  egnn_proj_kernel<<<NN, 128, 0, stream>>>(node_features, node_proj_w, node_proj_b, xbuf);
  egnn_copy_kernel<<<(NN * 2 + 255) / 256, 256, 0, stream>>>(pcur, pos, NN * 2);

  const int edge_smem = 98304 + 17408 + 5 * 64 * 4;  // 116992 B
  const int node_smem = 98304 + 17408;               // 115712 B
  for (int l = 0; l < LL; ++l) {
    egnn_zero_kernel<<<((size_t)NN * DD + 255) / 256, 256, 0, stream>>>(agg, NN * DD);
    egnn_zero_kernel<<<(NN * 2 + 255) / 256, 256, 0, stream>>>(pacc, NN * 2);
    egnn_edge_kernel<<<(NE + 63) / 64, 128, edge_smem, stream>>>(
        xbuf, pcur, ei, wep[l], w1last[l],
        edge_b1 + (size_t)l * DD, edge_b2 + (size_t)l * DD,
        pos_w1 + (size_t)l * (DD + 1), pos_b1 + l, pos_w2 + l, pos_b2 + l,
        agg, pacc);
    egnn_node_kernel<<<(NN + 63) / 64, 128, node_smem, stream>>>(
        xbuf, agg, wnp[l], node_b1 + (size_t)l * DD, node_b2 + (size_t)l * DD);
    egnn_axpy_kernel<<<(NN * 2 + 255) / 256, 256, 0, stream>>>(pcur, pacc, NN * 2);
  }

  egnn_ln_kernel<<<NN, 128, 0, stream>>>(xbuf, ln_g, ln_b, (float*)d_out);
}
